// VQBeTHead_69681549410998
// MI455X (gfx1250) — compile-verified, matrix operand order checked by reading, not run on hardware
//
#include <hip/hip_runtime.h>
#include <hip/hip_bf16.h>

// ---------------- problem dims ----------------
#define NT_    8192      // N*T
#define DIN_   2048
#define H_     2048
#define C_     16
#define G_     2
#define D_     256
#define WA_    35
#define GCWA_  1120      // G*C*WA
#define DECH_  512

typedef __attribute__((ext_vector_type(16))) __bf16 v16bf;
typedef __attribute__((ext_vector_type(8)))  __bf16 bf16x8;
typedef __attribute__((ext_vector_type(4)))  __bf16 bf16x4;
typedef __attribute__((ext_vector_type(8)))  float  v8f;

// B-tile load modes
#define NM_FULL   0   // N % 128 == 0 : no guards at all
#define NM_VEC    1   // N % 8   == 0 : octet-granular guards
#define NM_SCALAR 2   // arbitrary N  : per-element guards

__device__ __forceinline__ __bf16 f2bf(float f) {
  union { float f; unsigned u; } v; v.f = f;
  unsigned r = v.u + 0x7FFFu + ((v.u >> 16) & 1u);   // round-to-nearest-even
  unsigned short h = (unsigned short)(r >> 16);
  return __builtin_bit_cast(__bf16, h);
}

__device__ __forceinline__ v8f wmma_bf16(v16bf a, v16bf b, v8f c) {
#if defined(__HIP_DEVICE_COMPILE__)
  // D = A(16x32 bf16) * B(32x16 bf16) + C(16x16 f32)
  return __builtin_amdgcn_wmma_f32_16x16x32_bf16(
      /*neg_a=*/false, a, /*neg_b=*/false, b,
      /*c_mod=*/(short)0, c, /*reuse_a=*/false, /*reuse_b=*/false);
#else
  (void)a; (void)b; return c;
#endif
}

__device__ __forceinline__ void store_val(float* p, float v)  { *p = v; }
__device__ __forceinline__ void store_val(__bf16* p, float v) { *p = f2bf(v); }

// ---------------- fp32 -> bf16 convert (vectorized) ----------------
__global__ void cvt_f32_to_bf16(const float* __restrict__ src,
                                __bf16* __restrict__ dst, int n) {
  int tid = blockIdx.x * blockDim.x + threadIdx.x;
  int stride = gridDim.x * blockDim.x;
  int n4 = n >> 2;
  for (int i = tid; i < n4; i += stride) {
    float4 f = ((const float4*)src)[i];
    bf16x4 o;
    o[0] = f2bf(f.x); o[1] = f2bf(f.y); o[2] = f2bf(f.z); o[3] = f2bf(f.w);
    ((bf16x4*)dst)[i] = o;
  }
  for (int i = (n4 << 2) + tid; i < n; i += stride) dst[i] = f2bf(src[i]);
}

// ---------------- tiled bf16 WMMA GEMM, double-buffered LDS ----------------
// C(MxN) = act( A(MxK,bf16) @ B(KxN,bf16) + bias + optional selMat[rowSel[m]] )
// Block: 256 thr (8 waves). Tile 128x128, K-step 64, ping-pong LDS.
// Waves 4(M) x 2(N); each wave: 2x4 grid of 16x16 tiles -> 16 WMMA / stage.
// N, K are compile-time; global loads use running pointers (no in-loop muls).
// Requires: M % 128 == 0, K % 64 == 0.
template <typename OutT, bool RELU, bool HAS_SEL, int NMODE, int N, int K>
__global__ void __launch_bounds__(256)
gemm_bf16_wmma(const __bf16* __restrict__ A,
               const __bf16* __restrict__ B,
               const float*  __restrict__ bias,
               OutT* __restrict__ Cout,
               int M,
               const int*   __restrict__ rowSel,   // per-row selector (or null)
               const float* __restrict__ selMat) { // (16 x N) fp32 add-in
  constexpr int BM = 128, BN = 128, BK = 64, LDT = 72; // pad: 144B row stride
  __shared__ __attribute__((aligned(16))) __bf16 sA[2][BM * LDT]; // [row][k]
  __shared__ __attribute__((aligned(16))) __bf16 sB[2][BN * LDT]; // [col][k] transposed

  const int tid   = threadIdx.x;
  const int lane  = tid & 31;
  const int wave  = tid >> 5;
  const int waveM = wave & 3;   // 0..3 -> 32 rows each
  const int waveN = wave >> 2;  // 0..1 -> 64 cols each
  const int lrow  = lane & 15;  // row (A) / col (B,C) within 16x16 tile
  const int khalf = lane >> 4;  // selects K-octet per ISA 16-bit layout

  const int nBase = blockIdx.x * BN;
  const int mBase = blockIdx.y * BM;

  v8f acc[2][4];
  #pragma unroll
  for (int mi = 0; mi < 2; ++mi)
    #pragma unroll
    for (int ni = 0; ni < 4; ++ni) { v8f z = {}; acc[mi][ni] = z; }

  // ---- per-thread running global pointers (advance by constants) ----
  const __bf16* aP[4];
  const __bf16* bP[4];
  bool bOk[4];
  int  bCol0[4];
  #pragma unroll
  for (int it = 0; it < 4; ++it) {
    int c   = tid + it * 256;      // 0..1023
    int row = c >> 3;              // 0..127
    int seg = c & 7;               // 0..7 (8 bf16 each)
    aP[it] = A + (size_t)(mBase + row) * K + seg * 8;
  }
  #pragma unroll
  for (int it = 0; it < 4; ++it) {
    int c     = tid + it * 256;    // 0..1023
    int krow  = c >> 4;            // 0..63
    int cseg  = c & 15;            // 0..15
    int gcol0 = nBase + cseg * 8;
    bP[it]   = B + (size_t)krow * N + gcol0;
    bOk[it]  = gcol0 < N;          // exact when N%8==0
    bCol0[it] = gcol0;
  }

  bf16x8 ra[4], rb[4];
  auto loadAB = [&]() {
    #pragma unroll
    for (int it = 0; it < 4; ++it) {
      ra[it] = *(const bf16x8*)aP[it];
      aP[it] += BK;
    }
    #pragma unroll
    for (int it = 0; it < 4; ++it) {
      if (NMODE == NM_FULL) {
        rb[it] = *(const bf16x8*)bP[it];
      } else if (NMODE == NM_VEC) {
        bf16x8 v = {};
        if (bOk[it]) v = *(const bf16x8*)bP[it];
        rb[it] = v;
      } else {
        bf16x8 v = {};
        #pragma unroll
        for (int i = 0; i < 8; ++i)
          if (bCol0[it] + i < N) v[i] = bP[it][i];
        rb[it] = v;
      }
      bP[it] += (size_t)BK * N;
    }
  };
  auto stageLDS = [&](int p) {
    #pragma unroll
    for (int it = 0; it < 4; ++it) {
      int c   = tid + it * 256;
      int row = c >> 3;
      int seg = c & 7;
      *(bf16x8*)(&sA[p][row * LDT + seg * 8]) = ra[it];
    }
    #pragma unroll
    for (int it = 0; it < 4; ++it) {
      int c    = tid + it * 256;
      int krow = c >> 4;
      int cseg = c & 15;
      #pragma unroll
      for (int i = 0; i < 8; ++i)
        sB[p][(size_t)(cseg * 8 + i) * LDT + krow] = rb[it][i];
    }
  };
  auto compute = [&](int p) {
    #pragma unroll
    for (int kk = 0; kk < 2; ++kk) {     // two 32-wide K sub-steps
      const int kb = kk * 32 + khalf * 8;
      v16bf aF[2], bF[4];
      #pragma unroll
      for (int mi = 0; mi < 2; ++mi) {
        const __bf16* ap = &sA[p][(size_t)(waveM * 32 + mi * 16 + lrow) * LDT + kb];
        bf16x8 lo = *(const bf16x8*)ap;
        bf16x8 hi = *(const bf16x8*)(ap + 16);
        #pragma unroll
        for (int i = 0; i < 8; ++i) { aF[mi][i] = lo[i]; aF[mi][8 + i] = hi[i]; }
      }
      #pragma unroll
      for (int ni = 0; ni < 4; ++ni) {
        const __bf16* bp = &sB[p][(size_t)(waveN * 64 + ni * 16 + lrow) * LDT + kb];
        bf16x8 lo = *(const bf16x8*)bp;
        bf16x8 hi = *(const bf16x8*)(bp + 16);
        #pragma unroll
        for (int i = 0; i < 8; ++i) { bF[ni][i] = lo[i]; bF[ni][8 + i] = hi[i]; }
      }
      #pragma unroll
      for (int mi = 0; mi < 2; ++mi)
        #pragma unroll
        for (int ni = 0; ni < 4; ++ni)
          acc[mi][ni] = wmma_bf16(aF[mi], bF[ni], acc[mi][ni]);
    }
  };

  // ---- software pipeline: stage 0, then overlap loads with WMMA ----
  loadAB();
  stageLDS(0);
  __syncthreads();
  int p = 0;
  #pragma unroll 1
  for (int k0 = BK; k0 < K; k0 += BK) {
    loadAB();               // global loads in flight during compute
    compute(p);
    stageLDS(p ^ 1);        // consume loaded regs into the idle buffer
    __syncthreads();
    p ^= 1;
  }
  compute(p);

  // ---- epilogue: C layout VGPR r -> row = r + 8*khalf, col = lrow ----
  #pragma unroll
  for (int mi = 0; mi < 2; ++mi) {
    #pragma unroll
    for (int ni = 0; ni < 4; ++ni) {
      int col = nBase + waveN * 64 + ni * 16 + lrow;
      if (NMODE != NM_FULL && col >= N) continue;
      int rbase = mBase + waveM * 32 + mi * 16 + khalf * 8;
      float bv = bias ? bias[col] : 0.f;
      #pragma unroll
      for (int r = 0; r < 8; ++r) {
        int row = rbase + r;
        float v = acc[mi][ni][r] + bv;
        if (HAS_SEL) v += selMat[(size_t)rowSel[row] * N + col];
        if (RELU)    v = v > 0.f ? v : 0.f;
        store_val(&Cout[(size_t)row * N + col], v);
      }
    }
  }
  (void)M;
}

// ---------------- sampling: deterministic Gumbel-max categorical ----------------
__device__ __forceinline__ unsigned hashu(unsigned a) {
  a ^= a >> 16; a *= 0x7feb352du; a ^= a >> 15; a *= 0x846ca68bu; a ^= a >> 16;
  return a;
}

__global__ void categorical_kernel(const float* __restrict__ logits,
                                   int* __restrict__ sel,
                                   int rows, int cols, unsigned seed) {
  int r = blockIdx.x * blockDim.x + threadIdx.x;
  if (r >= rows) return;
  float best = -3.0e38f; int bi = 0;
  for (int i = 0; i < cols; ++i) {
    unsigned h = hashu(((unsigned)r * 2654435761u + (unsigned)i * 40503u) ^ seed);
    float u = ((float)h + 0.5f) * (1.0f / 4294967296.0f);
    float g = -__logf(-__logf(u));
    float s = logits[(size_t)r * cols + i] + g;
    if (s > best) { best = s; bi = i; }
  }
  sel[r] = bi;
}

// ---------------- codebook gather -> bf16 decoder input ----------------
__global__ void gather_codes_kernel(const float* __restrict__ codebook, // (G,C,D)
                                    const int* __restrict__ s1,
                                    const int* __restrict__ s2,
                                    __bf16* __restrict__ dec_in) {      // (NT, G*D)
  int i = blockIdx.x * blockDim.x + threadIdx.x;
  if (i >= NT_ * G_ * D_) return;
  int r = i >> 9;        // / 512
  int j = i & 511;
  int g = j >> 8;        // / 256
  int d = j & 255;
  int selv = g ? s2[r] : s1[r];
  float v = codebook[(size_t)(g * C_ + selv) * D_ + d];
  dec_in[i] = f2bf(v);
}

// ---------------- finalize: offset gather/sum + decoded add + centers ----------------
__global__ void finalize_kernel(const float* __restrict__ offs,     // (NT, G, C, WA)
                                const float* __restrict__ decoded,  // (NT, WA)
                                const int* __restrict__ s1,
                                const int* __restrict__ s2,
                                float* __restrict__ pred,           // (NT, WA)
                                float* __restrict__ centers) {      // (NT, 2)
  int i = blockIdx.x * blockDim.x + threadIdx.x;
  if (i >= NT_ * WA_) return;
  int r = i / WA_;
  int c = i - r * WA_;
  int a = s1[r], b = s2[r];
  float off = offs[(size_t)r * GCWA_ + a * WA_ + c]
            + offs[(size_t)r * GCWA_ + C_ * WA_ + b * WA_ + c];
  pred[i] = decoded[i] + off;
  if (c == 0) centers[r * 2 + 0] = (float)a;
  if (c == 1) centers[r * 2 + 1] = (float)b;
}

// ---------------- launch ----------------
extern "C" void kernel_launch(void* const* d_in, const int* in_sizes, int n_in,
                              void* d_out, int out_size, void* d_ws, size_t ws_size,
                              hipStream_t stream) {
  (void)in_sizes; (void)n_in; (void)out_size; (void)ws_size;
  const float* x   = (const float*)d_in[0];
  const float* w11 = (const float*)d_in[1];
  const float* b11 = (const float*)d_in[2];
  const float* w12 = (const float*)d_in[3];
  const float* b12 = (const float*)d_in[4];
  const float* w13 = (const float*)d_in[5];
  const float* b13 = (const float*)d_in[6];
  const float* w21 = (const float*)d_in[7];
  const float* b21 = (const float*)d_in[8];
  const float* w22 = (const float*)d_in[9];
  const float* b22 = (const float*)d_in[10];
  const float* wo1 = (const float*)d_in[11];
  const float* bo1 = (const float*)d_in[12];
  const float* wo2 = (const float*)d_in[13];
  const float* bo2 = (const float*)d_in[14];
  const float* wo3 = (const float*)d_in[15];
  const float* bo3 = (const float*)d_in[16];
  const float* codebook = (const float*)d_in[17];
  const float* wd1 = (const float*)d_in[18];
  const float* bd1 = (const float*)d_in[19];
  const float* wd2 = (const float*)d_in[20];
  const float* bd2 = (const float*)d_in[21];

  // outputs (concat, fp32): predicted, decoded, logits1, logits2, centers
  float* out_pred    = (float*)d_out;
  float* out_decoded = out_pred    + (size_t)NT_ * WA_;
  float* out_lg1     = out_decoded + (size_t)NT_ * WA_;
  float* out_lg2     = out_lg1     + (size_t)NT_ * C_;
  float* out_cent    = out_lg2     + (size_t)NT_ * C_;

  // workspace carve (256B aligned)
  char* ws = (char*)d_ws;
  size_t off = 0;
  auto carve = [&](size_t bytes) -> char* {
    char* p = ws + off;
    off = (off + bytes + 255) & ~(size_t)255;
    return p;
  };
  __bf16* Xb   = (__bf16*)carve((size_t)NT_ * DIN_ * 2);
  __bf16* w11b = (__bf16*)carve((size_t)DIN_ * H_ * 2);
  __bf16* w12b = (__bf16*)carve((size_t)H_ * H_ * 2);
  __bf16* w13b = (__bf16*)carve((size_t)H_ * C_ * 2);
  __bf16* w21b = (__bf16*)carve((size_t)DIN_ * H_ * 2);  // first 2048 rows of w21
  __bf16* w22b = (__bf16*)carve((size_t)H_ * C_ * 2);
  __bf16* wo1b = (__bf16*)carve((size_t)DIN_ * H_ * 2);
  __bf16* wo2b = (__bf16*)carve((size_t)H_ * H_ * 2);
  __bf16* wo3b = (__bf16*)carve((size_t)H_ * GCWA_ * 2);
  __bf16* wd1b = (__bf16*)carve((size_t)(G_ * D_) * DECH_ * 2);
  __bf16* wd2b = (__bf16*)carve((size_t)DECH_ * WA_ * 2);
  __bf16* bufA = (__bf16*)carve((size_t)NT_ * H_ * 2);
  __bf16* bufB = (__bf16*)carve((size_t)NT_ * H_ * 2);
  float*  offsb = (float*)carve((size_t)NT_ * GCWA_ * 4);
  int*    s1    = (int*)carve((size_t)NT_ * 4);
  int*    s2    = (int*)carve((size_t)NT_ * 4);
  __bf16* dec_in = (__bf16*)carve((size_t)NT_ * G_ * D_ * 2);
  __bf16* d1h    = (__bf16*)carve((size_t)NT_ * DECH_ * 2);

  auto cvt = [&](const float* s, __bf16* d, size_t n) {
    int blocks = (int)((n / 4 + 255) / 256); if (blocks > 4096) blocks = 4096;
    if (blocks < 1) blocks = 1;
    cvt_f32_to_bf16<<<blocks, 256, 0, stream>>>(s, d, (int)n);
  };
  cvt(x,   Xb,   (size_t)NT_ * DIN_);
  cvt(w11, w11b, (size_t)DIN_ * H_);
  cvt(w12, w12b, (size_t)H_ * H_);
  cvt(w13, w13b, (size_t)H_ * C_);
  cvt(w21, w21b, (size_t)DIN_ * H_);        // rows 0..2047 are contiguous
  cvt(w22, w22b, (size_t)H_ * C_);
  cvt(wo1, wo1b, (size_t)DIN_ * H_);
  cvt(wo2, wo2b, (size_t)H_ * H_);
  cvt(wo3, wo3b, (size_t)H_ * GCWA_);
  cvt(wd1, wd1b, (size_t)(G_ * D_) * DECH_);
  cvt(wd2, wd2b, (size_t)DECH_ * WA_);

  const dim3 blk(256);
  auto grid = [](int N, int M) { return dim3((unsigned)((N + 127) / 128), (unsigned)(M / 128)); };

  // bin1: h1 = relu(X@w11+b11); h2 = relu(h1@w12+b12); logits1 = h2@w13+b13
  gemm_bf16_wmma<__bf16, true, false, NM_FULL, H_, DIN_><<<grid(H_, NT_), blk, 0, stream>>>(
      Xb, w11b, b11, bufA, NT_, nullptr, nullptr);
  gemm_bf16_wmma<__bf16, true, false, NM_FULL, H_, H_><<<grid(H_, NT_), blk, 0, stream>>>(
      bufA, w12b, b12, bufB, NT_, nullptr, nullptr);
  gemm_bf16_wmma<float, false, false, NM_VEC, C_, H_><<<grid(C_, NT_), blk, 0, stream>>>(
      bufB, w13b, b13, out_lg1, NT_, nullptr, nullptr);

  // offsets: o1 = relu(X@wo1); o2 = relu(o1@wo2); offs = o2@wo3+bo3
  gemm_bf16_wmma<__bf16, true, false, NM_FULL, H_, DIN_><<<grid(H_, NT_), blk, 0, stream>>>(
      Xb, wo1b, bo1, bufA, NT_, nullptr, nullptr);
  gemm_bf16_wmma<__bf16, true, false, NM_FULL, H_, H_><<<grid(H_, NT_), blk, 0, stream>>>(
      bufA, wo2b, bo2, bufB, NT_, nullptr, nullptr);
  gemm_bf16_wmma<float, false, false, NM_VEC, GCWA_, H_><<<grid(GCWA_, NT_), blk, 0, stream>>>(
      bufB, wo3b, bo3, offsb, NT_, nullptr, nullptr);

  // sample s1
  categorical_kernel<<<(NT_ + 255) / 256, blk, 0, stream>>>(out_lg1, s1, NT_, C_, 0x9E3779B9u);

  // bin2: h = relu(X@w21[:2048] + w21[2048+s1] + b21); logits2 = h@w22+b22
  const float* w21tail = w21 + (size_t)DIN_ * H_;   // (16 x 2048) fp32
  gemm_bf16_wmma<__bf16, true, true, NM_FULL, H_, DIN_><<<grid(H_, NT_), blk, 0, stream>>>(
      Xb, w21b, b21, bufA, NT_, s1, w21tail);
  gemm_bf16_wmma<float, false, false, NM_VEC, C_, H_><<<grid(C_, NT_), blk, 0, stream>>>(
      bufA, w22b, b22, out_lg2, NT_, nullptr, nullptr);

  // sample s2
  categorical_kernel<<<(NT_ + 255) / 256, blk, 0, stream>>>(out_lg2, s2, NT_, C_, 0x85EBCA6Bu);

  // decoder: dec_in = gather(codebook); d1 = relu(dec_in@wd1+bd1); decoded = d1@wd2+bd2
  gather_codes_kernel<<<(NT_ * G_ * D_ + 255) / 256, blk, 0, stream>>>(codebook, s1, s2, dec_in);
  gemm_bf16_wmma<__bf16, true, false, NM_FULL, DECH_, G_ * D_><<<grid(DECH_, NT_), blk, 0, stream>>>(
      dec_in, wd1b, bd1, d1h, NT_, nullptr, nullptr);
  gemm_bf16_wmma<float, false, false, NM_SCALAR, WA_, DECH_><<<grid(WA_, NT_), blk, 0, stream>>>(
      d1h, wd2b, bd2, out_decoded, NT_, nullptr, nullptr);

  // final: predicted = decoded + gathered offsets; emit centers
  finalize_kernel<<<(NT_ * WA_ + 255) / 256, blk, 0, stream>>>(
      offsb, out_decoded, s1, s2, out_pred, out_cent);
}
